// GRUSimpleModel_91130616087183
// MI455X (gfx1250) — matrix operand order repaired and checked by reading, not compiled, not run on hardware
//
#include <hip/hip_runtime.h>

typedef __attribute__((ext_vector_type(16))) __bf16 v16bf;
typedef __attribute__((ext_vector_type(8)))  __bf16 bf16x8;
typedef __attribute__((ext_vector_type(8)))  float  v8f;

#define B_   256
#define T_   512
#define C_   37
#define U_   512
#define F_   111
#define FPAD 128

// ---- workspace layout (bytes) ----
#define WS_XBF  ((size_t)0)                               // [B*T][128] bf16  = 33.55 MB
#define WS_KB   (WS_XBF + (size_t)B_ * T_ * FPAD * 2)     // [3U][128]  bf16 (transposed kernel)
#define WS_RKB  (WS_KB  + (size_t)3 * U_ * FPAD * 2)      // [3U][512]  bf16 (transposed rkernel)
#define WS_A1   (WS_RKB + (size_t)3 * U_ * U_ * 2)        // [B][U] f32
#define WS_H0   (WS_A1  + (size_t)B_ * U_ * 4)            // [B][U] f32

// -------------------------------------------------------------------------
// Pack x = concat(values, measurements, dt) -> bf16, K padded 111 -> 128
// -------------------------------------------------------------------------
__global__ void k_prep_x(const float* __restrict__ values,
                         const int*   __restrict__ meas,
                         const float* __restrict__ dt,
                         __bf16*      __restrict__ xbf) {
    size_t idx = (size_t)blockIdx.x * blockDim.x + threadIdx.x;
    size_t bt  = idx >> 7;            // / 128
    int    f   = (int)(idx & 127);
    size_t base = bt * C_;
    float v = 0.f;
    if (f < C_)            v = values[base + f];
    else if (f < 2 * C_)   v = (float)meas[base + (f - C_)];
    else if (f < 3 * C_)   v = dt[base + (f - 2 * C_)];
    xbf[idx] = (__bf16)v;
}

// -------------------------------------------------------------------------
// Transpose + convert weights: kb[n][k] (K pad->128), rkb[n][k]
// -------------------------------------------------------------------------
__global__ void k_prep_w(const float* __restrict__ kern,
                         const float* __restrict__ rkern,
                         __bf16* __restrict__ kb,
                         __bf16* __restrict__ rkb) {
    int idx = blockIdx.x * blockDim.x + threadIdx.x;
    const int NKB = 3 * U_ * FPAD;
    if (idx < NKB) {
        int n = idx >> 7, k = idx & 127;
        kb[idx] = (k < F_) ? (__bf16)kern[(size_t)k * (3 * U_) + n] : (__bf16)0.f;
    } else {
        int j = idx - NKB;
        int n = j >> 9, k = j & 511;
        rkb[j] = (__bf16)rkern[(size_t)k * (3 * U_) + n];
    }
}

// -------------------------------------------------------------------------
// Demo encoder: a1 = relu(demo @ W_d1 + b_d1) ; h0 = a1 @ W_d2 + b_d2
// -------------------------------------------------------------------------
__global__ void k_h0a(const float* __restrict__ demo, const float* __restrict__ W,
                      const float* __restrict__ b, float* __restrict__ a1) {
    int idx = blockIdx.x * blockDim.x + threadIdx.x;
    int bi = idx >> 9, u = idx & 511;
    float s = b[u];
#pragma unroll
    for (int k = 0; k < 16; ++k) s += demo[bi * 16 + k] * W[k * U_ + u];
    a1[idx] = fmaxf(s, 0.f);
}

__global__ void k_h0b(const float* __restrict__ a1, const float* __restrict__ W,
                      const float* __restrict__ b, float* __restrict__ h0) {
    int idx = blockIdx.x * blockDim.x + threadIdx.x;
    int bi = idx >> 9, u = idx & 511;
    float s = b[u];
    for (int k = 0; k < U_; ++k) s += a1[bi * U_ + k] * W[k * U_ + u];
    h0[idx] = s;
}

// -------------------------------------------------------------------------
// WMMA helpers
// -------------------------------------------------------------------------
__device__ inline v8f wmma_bf16(v16bf a, v16bf b, v8f c) {
    return __builtin_amdgcn_wmma_f32_16x16x32_bf16(false, a, false, b,
                                                   (short)0, c, false, false);
}

// B-fragment (32x16 bf16): lane holds column `ncol`, 16 K values
// (lanes 0-15 -> K = kc*32+0..7,16..23, lanes 16-31 -> K = kc*32+8..15,24..31).
// Weights stored transposed [N][K] so this is two contiguous 16B loads.
__device__ inline v16bf load_bfrag(const __bf16* base, int rowstride,
                                   int ncol, int kc, int kh) {
    union { v16bf v; bf16x8 h[2]; } b;
    const __bf16* p = base + (size_t)ncol * rowstride + kc * 32 + kh * 8;
    b.h[0] = *(const bf16x8*)p;
    b.h[1] = *(const bf16x8*)(p + 16);
    return b.v;
}

// -------------------------------------------------------------------------
// GRU: 16 blocks x 512 threads (16 waves). Block owns 16 batch rows.
// Per step, wave w computes 2 u-tiles (16 cols each) of the 3 gate GEMMs.
// NOTE: xbf/kb/rkb intentionally NOT __restrict__: combined with the asm
// memory clobber in the time loop this stops LLVM from hoisting the
// loop-invariant weight-fragment loads (which previously spilled to scratch).
// -------------------------------------------------------------------------
__launch_bounds__(512)
__global__ void k_gru(const __bf16* xbf,
                      const __bf16* kb,
                      const __bf16* rkb,
                      const float*  __restrict__ h0,
                      const float*  __restrict__ bias,
                      const long long* __restrict__ lengths,
                      const float*  __restrict__ W_out,
                      const float*  __restrict__ b_out,
                      float*        __restrict__ out) {
    __shared__ __align__(16) float  h_f[16][U_ + 4];   // fp32 state
    __shared__ __align__(16) __bf16 h_b[16][U_ + 8];   // bf16 copy for A frags
    __shared__ int len_s[16];

    const int tid  = threadIdx.x;
    const int lane = tid & 31;
    const int wave = tid >> 5;
    const int m0   = blockIdx.x * 16;
    const int mr   = lane & 15;   // row (A) / column-within-tile (B/C/D)
    const int kh   = lane >> 4;   // lane-half selector

    // load h0 tile into LDS
    for (int i = tid; i < 16 * U_; i += 512) {
        int m = i >> 9, k = i & (U_ - 1);
        float v = h0[(size_t)(m0 + m) * U_ + k];
        h_f[m][k] = v;
        h_b[m][k] = (__bf16)v;
    }
    if (tid < 16) len_s[tid] = (int)lengths[m0 + tid];

    const float* b0 = bias;            // bias[0]
    const float* b1 = bias + 3 * U_;   // bias[1] (recurrent)

    // per-lane constants for this wave's two u-tiles
    float bz[2], br[2], bxh[2], brh[2];
    int   ucol[2];
#pragma unroll
    for (int ut = 0; ut < 2; ++ut) {
        int u = (wave * 2 + ut) * 16 + mr;
        ucol[ut] = u;
        bz[ut]  = b0[u] + b1[u];
        br[ut]  = b0[U_ + u] + b1[U_ + u];
        bxh[ut] = b0[2 * U_ + u];
        brh[ut] = b1[2 * U_ + u];
    }

    const __bf16* xrow = xbf + (size_t)(m0 + mr) * T_ * FPAD;

    __syncthreads();

    for (int t = 0; t < T_; ++t) {
        // Compiler memory barrier: forbid hoisting weight loads out of the
        // time loop (they must stream from L2 each step, not spill to scratch).
        asm volatile("" ::: "memory");

        v8f az[2] = {}, ar[2] = {}, axh[2] = {}, arh[2] = {};

        // ---- input projection: A = x[:, t, :]  (K = 128) ----
        const __bf16* xp = xrow + (size_t)t * FPAD;
#pragma unroll
        for (int kc = 0; kc < 4; ++kc) {
            union { v16bf v; bf16x8 h[2]; } a;
            a.h[0] = *(const bf16x8*)(xp + kc * 32 + kh * 8);
            a.h[1] = *(const bf16x8*)(xp + kc * 32 + 16 + kh * 8);
#pragma unroll
            for (int ut = 0; ut < 2; ++ut) {
                int nc = (wave * 2 + ut) * 16 + mr;
                az[ut]  = wmma_bf16(a.v, load_bfrag(kb, FPAD, nc,           kc, kh), az[ut]);
                ar[ut]  = wmma_bf16(a.v, load_bfrag(kb, FPAD, nc + U_,      kc, kh), ar[ut]);
                axh[ut] = wmma_bf16(a.v, load_bfrag(kb, FPAD, nc + 2 * U_,  kc, kh), axh[ut]);
            }
        }

        // ---- recurrent projection: A = h (K = 512) ----
#pragma unroll 4
        for (int kc = 0; kc < 16; ++kc) {
            union { v16bf v; bf16x8 h[2]; } a;
            const __bf16* hp = &h_b[mr][kc * 32 + kh * 8];
            a.h[0] = *(const bf16x8*)hp;
            a.h[1] = *(const bf16x8*)(hp + 16);
#pragma unroll
            for (int ut = 0; ut < 2; ++ut) {
                int nc = (wave * 2 + ut) * 16 + mr;
                az[ut]  = wmma_bf16(a.v, load_bfrag(rkb, U_, nc,          kc, kh), az[ut]);
                ar[ut]  = wmma_bf16(a.v, load_bfrag(rkb, U_, nc + U_,     kc, kh), ar[ut]);
                arh[ut] = wmma_bf16(a.v, load_bfrag(rkb, U_, nc + 2 * U_, kc, kh), arh[ut]);
            }
        }

        __syncthreads();   // all reads of h_b done before overwrite

        // ---- gate math + state update (C/D layout: VGPR i -> M=i+kh*8, N=lane%16) ----
#pragma unroll
        for (int ut = 0; ut < 2; ++ut) {
            int n = ucol[ut];
#pragma unroll
            for (int i = 0; i < 8; ++i) {
                int m  = i + kh * 8;
                float z  = 1.f / (1.f + __expf(-(az[ut][i] + bz[ut])));
                float r  = 1.f / (1.f + __expf(-(ar[ut][i] + br[ut])));
                float hh = tanhf(axh[ut][i] + bxh[ut] + r * (arh[ut][i] + brh[ut]));
                float ho = h_f[m][n];
                float hn = z * ho + (1.f - z) * hh;
                if (t >= len_s[m]) hn = ho;
                h_f[m][n] = hn;
                h_b[m][n] = (__bf16)hn;
            }
        }
        __syncthreads();   // writes visible before next step's reads
    }

    // ---- output head: out = sigmoid(h @ W_out + b_out), one wave per row ----
    {
        int m = wave;      // 16 waves -> 16 rows
        float p = 0.f;
        for (int k = lane; k < U_; k += 32) p += h_f[m][k] * W_out[k];
#pragma unroll
        for (int off = 16; off > 0; off >>= 1) p += __shfl_down(p, off, 32);
        if (lane == 0) out[m0 + m] = 1.f / (1.f + __expf(-(p + b_out[0])));
    }
}

// -------------------------------------------------------------------------
extern "C" void kernel_launch(void* const* d_in, const int* in_sizes, int n_in,
                              void* d_out, int out_size, void* d_ws, size_t ws_size,
                              hipStream_t stream) {
    const float*     demo    = (const float*)d_in[0];
    const float*     values  = (const float*)d_in[1];
    const int*       meas    = (const int*)  d_in[2];
    const float*     dt      = (const float*)d_in[3];
    const long long* lengths = (const long long*)d_in[4];
    const float*     W_d1    = (const float*)d_in[5];
    const float*     b_d1    = (const float*)d_in[6];
    const float*     W_d2    = (const float*)d_in[7];
    const float*     b_d2    = (const float*)d_in[8];
    const float*     kern    = (const float*)d_in[9];
    const float*     rkern   = (const float*)d_in[10];
    const float*     bias    = (const float*)d_in[11];
    const float*     W_out   = (const float*)d_in[12];
    const float*     b_out   = (const float*)d_in[13];
    float* out = (float*)d_out;

    char*   ws  = (char*)d_ws;
    __bf16* xbf = (__bf16*)(ws + WS_XBF);
    __bf16* kb  = (__bf16*)(ws + WS_KB);
    __bf16* rkb = (__bf16*)(ws + WS_RKB);
    float*  a1  = (float*) (ws + WS_A1);
    float*  h0  = (float*) (ws + WS_H0);

    k_prep_x<<<(B_ * T_ * FPAD) / 256, 256, 0, stream>>>(values, meas, dt, xbf);
    k_prep_w<<<(3 * U_ * (FPAD + U_)) / 256, 256, 0, stream>>>(kern, rkern, kb, rkb);
    k_h0a<<<(B_ * U_) / 256, 256, 0, stream>>>(demo, W_d1, b_d1, a1);
    k_h0b<<<(B_ * U_) / 256, 256, 0, stream>>>(a1, W_d2, b_d2, h0);
    k_gru<<<B_ / 16, 512, 0, stream>>>(xbf, kb, rkb, h0, bias, lengths,
                                       W_out, b_out, out);
}
